// BertAttention_90297392431744
// MI455X (gfx1250) — matrix operand order repaired and checked by reading, not compiled, not run on hardware
//
#include <hip/hip_runtime.h>
#include <hip/hip_bf16.h>

typedef __bf16 bf16;
typedef __attribute__((ext_vector_type(16))) __bf16 v16bf;
typedef __attribute__((ext_vector_type(8)))  float  v8f;
typedef __attribute__((ext_vector_type(4)))  unsigned int u32x4;
typedef __attribute__((ext_vector_type(8)))  unsigned int u32x8;
typedef __attribute__((ext_vector_type(4)))  float  f32x4;

constexpr int kB  = 4;
constexpr int kS  = 1024;
constexpr int kH  = 768;
constexpr int kNH = 12;
constexpr int kDH = 64;
constexpr float kNEG = -100000.0f;

// ---------- WMMA helpers (CDNA5 16x16x32 bf16, fp32 accumulate) ----------

__device__ inline v16bf mk_frag(u32x4 a, u32x4 b) {
  u32x8 t;
  t[0] = a[0]; t[1] = a[1]; t[2] = a[2]; t[3] = a[3];
  t[4] = b[0]; t[5] = b[1]; t[6] = b[2]; t[7] = b[3];
  return __builtin_bit_cast(v16bf, t);
}

// A fragment: 16x32 bf16, memory row-major (row stride = ld halfs).
// Lane l: row = l&15; chunks at cols (l>>4)*8 and 16+(l>>4)*8 (each 16B).
__device__ inline v16bf load_a16x32(const bf16* __restrict__ p, int ld) {
  const int lane = threadIdx.x & 31;
  const int row = lane & 15, hf = lane >> 4;
  const bf16* q = p + row * ld + hf * 8;
  u32x4 c0 = *(const u32x4*)(q);
  u32x4 c1 = *(const u32x4*)(q + 16);
  return mk_frag(c0, c1);
}

// B fragment: 32x16 bf16 where memory holds B^T (16 rows x 32 cols, row stride ld).
// Lane l: col N = l&15; chunks at K (l>>4)*8 and 16+(l>>4)*8 (each 16B).
__device__ inline v16bf load_bT32x16(const bf16* __restrict__ p, int ld) {
  const int lane = threadIdx.x & 31;
  const int n = lane & 15, hf = lane >> 4;
  const bf16* q = p + n * ld + hf * 8;
  u32x4 c0 = *(const u32x4*)(q);
  u32x4 c1 = *(const u32x4*)(q + 16);
  return mk_frag(c0, c1);
}

__device__ inline v8f wmma_bf16(v16bf a, v16bf b, v8f c) {
  return __builtin_amdgcn_wmma_f32_16x16x32_bf16(false, a, false, b, (short)0, c,
                                                 false, false);
}

__device__ inline unsigned pack2bf(float a, float b) {
  unsigned short lo = __builtin_bit_cast(unsigned short, (bf16)a);
  unsigned short hi = __builtin_bit_cast(unsigned short, (bf16)b);
  return (unsigned)lo | ((unsigned)hi << 16);
}

// ---------- prep: masked vis_position add + bf16 conversion ----------

__global__ void prep_act(const float* __restrict__ hidden,
                         const float* __restrict__ context,
                         const float* __restrict__ vis,
                         const int* __restrict__ feat_len,
                         bf16* __restrict__ hsb, bf16* __restrict__ ctxb) {
  int i = blockIdx.x * blockDim.x + threadIdx.x;
  if (i >= kB * kS * kH) return;
  int b = i / (kS * kH);
  int s = (i / kH) & (kS - 1);
  float add = (s < feat_len[b]) ? vis[i] : 0.0f;
  hsb[i]  = (bf16)(hidden[i]  + add);
  ctxb[i] = (bf16)(context[i] + add);
}

__global__ void prep_w(const float* __restrict__ Wq,
                       const float* __restrict__ Wk,
                       const float* __restrict__ Wv,
                       bf16* __restrict__ w) {
  int i = blockIdx.x * blockDim.x + threadIdx.x;
  if (i >= 3 * kH * kH) return;
  int off = i % (kH * kH);
  const float* src = (i < kH * kH) ? Wq : (i < 2 * kH * kH ? Wk : Wv);
  w[i] = (bf16)src[off];
}

// ---------- QKV projection GEMM: one wave = 16-row x 64-col tile ----------
// grid.x = 256 m-tiles * 12 n64-tiles, grid.y = which matrix (0=Q,1=K,2=V^T)
// Two-stage ping-pong pipeline: stage B's fragments load while stage A's
// WMMAs run, and vice versa -- no register rotation moves.

__global__ __launch_bounds__(32)
void qkv_kernel(const bf16* __restrict__ hsb, const bf16* __restrict__ ctxb,
                const bf16* __restrict__ wqkv,
                const float* __restrict__ bq, const float* __restrict__ bk,
                const float* __restrict__ bv,
                bf16* __restrict__ Qw, bf16* __restrict__ Kw,
                bf16* __restrict__ VTw) {
  const int which = blockIdx.y;
  const int mt = blockIdx.x / kNH;   // 16-row tile index over B*S rows
  const int nt = blockIdx.x % kNH;   // head index (64 output cols)
  const int lane = threadIdx.x & 31;

  const bf16* A = (which == 0 ? hsb : ctxb) + (size_t)mt * 16 * kH;
  const bf16* W = wqkv + (size_t)which * kH * kH + (size_t)nt * 64 * kH;
  const float* bias = (which == 0 ? bq : (which == 1 ? bk : bv)) + nt * 64;

  constexpr int KSTEPS = kH / 32;  // 24 (even)
  v8f acc[4] = {};
  v16bf afA, afB, wfA[4], wfB[4];

  afA = load_a16x32(A, kH);
#pragma unroll
  for (int j = 0; j < 4; ++j) wfA[j] = load_bT32x16(W + (size_t)j * 16 * kH, kH);

  for (int kk = 0; kk < KSTEPS; kk += 2) {
    const int k1 = (kk + 1) * 32;                             // kk+1 <= 23
    afB = load_a16x32(A + k1, kH);
#pragma unroll
    for (int j = 0; j < 4; ++j)
      wfB[j] = load_bT32x16(W + (size_t)j * 16 * kH + k1, kH);
#pragma unroll
    for (int j = 0; j < 4; ++j) acc[j] = wmma_bf16(afA, wfA[j], acc[j]);

    const int k2 = (kk + 2 < KSTEPS ? kk + 2 : KSTEPS - 1) * 32;  // clamped
    afA = load_a16x32(A + k2, kH);
#pragma unroll
    for (int j = 0; j < 4; ++j)
      wfA[j] = load_bT32x16(W + (size_t)j * 16 * kH + k2, kH);
#pragma unroll
    for (int j = 0; j < 4; ++j) acc[j] = wmma_bf16(afB, wfB[j], acc[j]);
  }

  const int n = lane & 15, hf = lane >> 4;
#pragma unroll
  for (int j = 0; j < 4; ++j) {
    float bb = bias[j * 16 + n];
    int d = j * 16 + n;  // 0..63 within head
    if (which < 2) {
      bf16* dst = (which == 0 ? Qw : Kw);
#pragma unroll
      for (int r = 0; r < 8; ++r) {
        int sg = mt * 16 + r + 8 * hf;          // global row in [0, B*S)
        int b = sg >> 10, s = sg & (kS - 1);
        dst[((((size_t)b * kNH + nt) << 10) + s) * kDH + d] =
            (bf16)(acc[j][r] + bb);
      }
    } else {
      // transposed store: VT[(b*NH+head)*64 + d][s], 8 consecutive s per lane
      int sg0 = mt * 16 + 8 * hf;
      int b = sg0 >> 10, s0 = sg0 & (kS - 1);
      bf16* dst = VTw + ((((size_t)b * kNH + nt) * kDH + d) << 10) + s0;
      u32x4 pk;
#pragma unroll
      for (int r = 0; r < 4; ++r)
        pk[r] = pack2bf(acc[j][2 * r] + bb, acc[j][2 * r + 1] + bb);
      *(u32x4*)dst = pk;
    }
  }
}

// ---------- attention: one wave = (b, head, 16-row q-tile) ----------

__global__ __launch_bounds__(32)
void attn_kernel(const bf16* __restrict__ Qw, const bf16* __restrict__ Kw,
                 const bf16* __restrict__ VTw,
                 const float* __restrict__ amask, const float* __restrict__ gate,
                 const int* __restrict__ feat_len, const int* __restrict__ prox,
                 float* __restrict__ out, float* __restrict__ scores_out) {
  __shared__ float lds[16][1028];   // pitch 1028 dwords -> conflict-free tiles
  __shared__ float maxr[16], invs[16];

  const int lane = threadIdx.x & 31;
  const int qt = blockIdx.x & 63;
  const int bh = blockIdx.x >> 6;
  const int b = bh / kNH;
  const int hd = bh % kNH;
  const int q0 = qt * 16;
  const int pp = prox[0];
  const int fl = feat_len[b];
  const int n = lane & 15, hf = lane >> 4;

  const bf16* Qb = Qw + (size_t)bh * kS * kDH + (size_t)q0 * kDH;
  const bf16* Kb = Kw + (size_t)bh * kS * kDH;
  const bf16* Vb = VTw + (size_t)bh * kDH * kS;

  v16bf qf0 = load_a16x32(Qb, kDH);        // d = 0..31
  v16bf qf1 = load_a16x32(Qb + 32, kDH);   // d = 32..63

  const float scale = 0.125f;  // 1/sqrt(64)
  float* sc_base = scores_out + ((size_t)bh * kS + q0) * kS;
  const float* gate_b = gate + ((size_t)b * kS + q0) * kS;
  const float* am_b = amask + (size_t)b * kS;

  // Pass 1: scores = (Q K^T)*scale*gate + mask, with NEG masking; -> LDS + HBM.
  // Two-stage ping-pong on K fragments + gate/mask scalars.
  auto score_tile = [&](int kt, v16bf k0f, v16bf k1f, const float* gv,
                        float amv) {
    v8f c = {};
    c = wmma_bf16(qf0, k0f, c);
    c = wmma_bf16(qf1, k1f, c);
    const int col = kt * 16 + n;
#pragma unroll
    for (int r = 0; r < 8; ++r) {
      int m = r + 8 * hf;
      int row = q0 + m;
      float v = c[r] * scale * gv[r] + amv;
      bool blk  = (row >= pp) && (col >= pp);
      bool band = (row < pp - 1) && ((col > row) || (col < row - 1));
      bool m3   = (row >= pp) && (col < fl - 1);
      if (blk || band || m3) v = kNEG;
      lds[m][col] = v;
      sc_base[(size_t)m * kS + col] = v;
    }
  };

  v16bf kf0A, kf1A, kf0B, kf1B;
  float gvA[8], gvB[8], amvA, amvB;

  kf0A = load_bT32x16(Kb, kDH);
  kf1A = load_bT32x16(Kb + 32, kDH);
  amvA = am_b[n];
#pragma unroll
  for (int r = 0; r < 8; ++r) gvA[r] = gate_b[(size_t)(r + 8 * hf) * kS + n];

  for (int kt = 0; kt < 64; kt += 2) {
    {  // prefetch tile kt+1 into B (kt+1 <= 63 always)
      const int t = kt + 1;
      kf0B = load_bT32x16(Kb + (size_t)t * 16 * kDH, kDH);
      kf1B = load_bT32x16(Kb + (size_t)t * 16 * kDH + 32, kDH);
      const int col = t * 16 + n;
      amvB = am_b[col];
#pragma unroll
      for (int r = 0; r < 8; ++r)
        gvB[r] = gate_b[(size_t)(r + 8 * hf) * kS + col];
    }
    score_tile(kt, kf0A, kf1A, gvA, amvA);
    {  // prefetch tile kt+2 into A (clamped at 63)
      const int t = (kt + 2 < 64) ? kt + 2 : 63;
      kf0A = load_bT32x16(Kb + (size_t)t * 16 * kDH, kDH);
      kf1A = load_bT32x16(Kb + (size_t)t * 16 * kDH + 32, kDH);
      const int col = t * 16 + n;
      amvA = am_b[col];
#pragma unroll
      for (int r = 0; r < 8; ++r)
        gvA[r] = gate_b[(size_t)(r + 8 * hf) * kS + col];
    }
    score_tile(kt + 1, kf0B, kf1B, gvB, amvB);
  }
  __syncthreads();

  // Row-wise softmax statistics (one lane per row, float4 LDS reads)
  if (lane < 16) {
    const float* rowp = &lds[lane][0];
    float mx = -3.0e38f;
    for (int c4 = 0; c4 < kS; c4 += 4) {
      f32x4 v = *(const f32x4*)(rowp + c4);
      mx = fmaxf(fmaxf(fmaxf(mx, v[0]), v[1]), fmaxf(v[2], v[3]));
    }
    float sum = 0.0f;
    for (int c4 = 0; c4 < kS; c4 += 4) {
      f32x4 v = *(const f32x4*)(rowp + c4);
      sum += __expf(v[0] - mx) + __expf(v[1] - mx) +
             __expf(v[2] - mx) + __expf(v[3] - mx);
    }
    maxr[lane] = mx;
    invs[lane] = 1.0f / sum;
  }
  __syncthreads();

  // Pass 2: out = softmax(scores) @ V, two-stage ping-pong on V^T fragments;
  // the LDS reads + exp math that build the prob frag overlap the loads.
  const float mrow = maxr[n];
  const float irow = invs[n];
  v8f oacc[4] = {};

  auto pv_step = [&](int kc, const v16bf* vf) {
    const float* rp = &lds[n][kc * 32 + hf * 8];
    f32x4 f0 = *(const f32x4*)(rp);
    f32x4 f1 = *(const f32x4*)(rp + 4);
    f32x4 f2 = *(const f32x4*)(rp + 16);
    f32x4 f3 = *(const f32x4*)(rp + 20);
    u32x8 t;
    t[0] = pack2bf(__expf(f0[0] - mrow) * irow, __expf(f0[1] - mrow) * irow);
    t[1] = pack2bf(__expf(f0[2] - mrow) * irow, __expf(f0[3] - mrow) * irow);
    t[2] = pack2bf(__expf(f1[0] - mrow) * irow, __expf(f1[1] - mrow) * irow);
    t[3] = pack2bf(__expf(f1[2] - mrow) * irow, __expf(f1[3] - mrow) * irow);
    t[4] = pack2bf(__expf(f2[0] - mrow) * irow, __expf(f2[1] - mrow) * irow);
    t[5] = pack2bf(__expf(f2[2] - mrow) * irow, __expf(f2[3] - mrow) * irow);
    t[6] = pack2bf(__expf(f3[0] - mrow) * irow, __expf(f3[1] - mrow) * irow);
    t[7] = pack2bf(__expf(f3[2] - mrow) * irow, __expf(f3[3] - mrow) * irow);
    v16bf pf = __builtin_bit_cast(v16bf, t);
#pragma unroll
    for (int dt = 0; dt < 4; ++dt) oacc[dt] = wmma_bf16(pf, vf[dt], oacc[dt]);
  };

  v16bf vfA[4], vfB[4];
#pragma unroll
  for (int dt = 0; dt < 4; ++dt)
    vfA[dt] = load_bT32x16(Vb + (size_t)(dt * 16) * kS, kS);

  for (int kc = 0; kc < 32; kc += 2) {
    const int k1 = (kc + 1) * 32;                              // kc+1 <= 31
#pragma unroll
    for (int dt = 0; dt < 4; ++dt)
      vfB[dt] = load_bT32x16(Vb + (size_t)(dt * 16) * kS + k1, kS);
    pv_step(kc, vfA);

    const int k2 = (kc + 2 < 32 ? kc + 2 : 31) * 32;           // clamped
#pragma unroll
    for (int dt = 0; dt < 4; ++dt)
      vfA[dt] = load_bT32x16(Vb + (size_t)(dt * 16) * kS + k2, kS);
    pv_step(kc + 1, vfB);
  }

  float* ob = out + ((size_t)b * kS + q0) * kH + hd * kDH;
#pragma unroll
  for (int dt = 0; dt < 4; ++dt) {
#pragma unroll
    for (int r = 0; r < 8; ++r) {
      int m = r + 8 * hf;
      ob[(size_t)m * kH + dt * 16 + n] = oacc[dt][r];
    }
  }
}

// ---------- launch ----------

extern "C" void kernel_launch(void* const* d_in, const int* in_sizes, int n_in,
                              void* d_out, int out_size, void* d_ws, size_t ws_size,
                              hipStream_t stream) {
  const float* hidden  = (const float*)d_in[0];
  const float* context = (const float*)d_in[1];
  const float* amask   = (const float*)d_in[2];
  const float* gate    = (const float*)d_in[3];
  const float* vis     = (const float*)d_in[4];
  const float* Wq      = (const float*)d_in[5];
  const float* bq      = (const float*)d_in[6];
  const float* Wk      = (const float*)d_in[7];
  const float* bk      = (const float*)d_in[8];
  const float* Wv      = (const float*)d_in[9];
  const float* bv      = (const float*)d_in[10];
  const int*   feat_len = (const int*)d_in[11];
  const int*   prox     = (const int*)d_in[12];

  float* out    = (float*)d_out;
  float* scores = out + (size_t)kB * kS * kH;  // tuple: (out, scores) flat

  char* ws = (char*)d_ws;
  const size_t act_sz = (size_t)kB * kS * kH * sizeof(bf16);
  const size_t w_sz   = (size_t)3 * kH * kH * sizeof(bf16);
  bf16* hsb  = (bf16*)(ws);                     ws += act_sz;
  bf16* ctxb = (bf16*)(ws);                     ws += act_sz;
  bf16* wqkv = (bf16*)(ws);                     ws += w_sz;
  bf16* Qw   = (bf16*)(ws);                     ws += act_sz;
  bf16* Kw   = (bf16*)(ws);                     ws += act_sz;
  bf16* VTw  = (bf16*)(ws);                     ws += act_sz;

  const int nAct = kB * kS * kH;
  prep_act<<<(nAct + 255) / 256, 256, 0, stream>>>(hidden, context, vis, feat_len,
                                                   hsb, ctxb);
  const int nW = 3 * kH * kH;
  prep_w<<<(nW + 255) / 256, 256, 0, stream>>>(Wq, Wk, Wv, wqkv);

  dim3 gQKV((kB * kS / 16) * kNH, 3);  // 3072 tiles x {Q,K,V}
  qkv_kernel<<<gQKV, 32, 0, stream>>>(hsb, ctxb, wqkv, bq, bk, bv, Qw, Kw, VTw);

  attn_kernel<<<kB * kNH * (kS / 16), 32, 0, stream>>>(Qw, Kw, VTw, amask, gate,
                                                       feat_len, prox, out, scores);
}
